// TrajectoryGenerator_85375359910302
// MI455X (gfx1250) — matrix operand order repaired and checked by reading, not compiled, not run on hardware
//
#include <hip/hip_runtime.h>
#include <hip/hip_bf16.h>

#define OBS_LEN 8
#define PRED_LEN 12
#define NAG 512
#define NUM_SEQ 8
#define H_ENC 8
#define H_DEC 16
#define EMB 16
#define NOISE_DIM 8

typedef __attribute__((ext_vector_type(16))) _Float16 v16h;
typedef __attribute__((ext_vector_type(8)))  float    v8f;

union VH { v16h v; unsigned u[8]; };

__device__ __forceinline__ float sigm(float x) { return 1.0f / (1.0f + __expf(-x)); }

__device__ __forceinline__ unsigned pack_f16x2(float a, float b) {
    union { _Float16 h[2]; unsigned u; } p;
    p.h[0] = (_Float16)a; p.h[1] = (_Float16)b;
    return p.u;
}

__device__ __forceinline__ float wave_max_f32(float v) {
    #pragma unroll
    for (int off = 16; off > 0; off >>= 1)
        v = fmaxf(v, __shfl_xor(v, off, 32));
    return v;
}
__device__ __forceinline__ float wave_sum_f32(float v) {
    #pragma unroll
    for (int off = 16; off > 0; off >>= 1)
        v += __shfl_xor(v, off, 32);
    return v;
}

// ---------------------------------------------------------------------------
// Convert pool-MLP weights to f16 once per launch (deterministic).
// Wm1h: [64][32] row-major (out,in), Wm2h: [16][64] row-major (out,in).
// ---------------------------------------------------------------------------
__global__ void conv_weights_kernel(const float* __restrict__ W_m1,
                                    const float* __restrict__ W_m2,
                                    _Float16* __restrict__ Wm1h,
                                    _Float16* __restrict__ Wm2h) {
    int i = blockIdx.x * blockDim.x + threadIdx.x;
    if (i < 64 * 32) Wm1h[i] = (_Float16)W_m1[i];
    if (i < 16 * 64) Wm2h[i] = (_Float16)W_m2[i];
}

// ---------------------------------------------------------------------------
// Encoder: per-agent tiny LSTM (H=8) over 8 obs steps, then concat noise and
// initialize all decoder state buffers.
// ---------------------------------------------------------------------------
__global__ void encoder_kernel(const float* __restrict__ traj_rel,
                               const float* __restrict__ obs_traj_pos,
                               const float* __restrict__ h0,
                               const float* __restrict__ c0,
                               const float* __restrict__ noise,
                               const float* __restrict__ W_enc, const float* __restrict__ b_enc,
                               const float* __restrict__ Wih_t, const float* __restrict__ Whh_t,
                               const float* __restrict__ bih_t, const float* __restrict__ bhh_t,
                               const int* __restrict__ seq_start_end,
                               float* __restrict__ hbuf, float* __restrict__ cbuf,
                               float* __restrict__ ctxbuf, float* __restrict__ outbuf,
                               float* __restrict__ posbuf) {
    int n = blockIdx.x * blockDim.x + threadIdx.x;
    if (n >= NAG) return;
    float h[H_ENC], c[H_ENC];
    #pragma unroll
    for (int k = 0; k < H_ENC; ++k) { h[k] = h0[n * H_ENC + k]; c[k] = c0[n * H_ENC + k]; }
    for (int t = 0; t < OBS_LEN; ++t) {
        float x0 = traj_rel[((size_t)t * NAG + n) * 2 + 0];
        float x1 = traj_rel[((size_t)t * NAG + n) * 2 + 1];
        float e[EMB];
        #pragma unroll
        for (int m = 0; m < EMB; ++m) {
            float v = W_enc[m * 2 + 0] * x0 + W_enc[m * 2 + 1] * x1 + b_enc[m];
            e[m] = v > 0.f ? v : 0.f;
        }
        float g[4 * H_ENC];
        #pragma unroll
        for (int q = 0; q < 4 * H_ENC; ++q) {
            float v = bih_t[q] + bhh_t[q];
            #pragma unroll
            for (int m = 0; m < EMB; ++m) v += Wih_t[q * EMB + m] * e[m];
            #pragma unroll
            for (int k = 0; k < H_ENC; ++k) v += Whh_t[q * H_ENC + k] * h[k];
            g[q] = v;
        }
        #pragma unroll
        for (int k = 0; k < H_ENC; ++k) {
            float ig = sigm(g[k]), fg = sigm(g[H_ENC + k]);
            float gg = tanhf(g[2 * H_ENC + k]), og = sigm(g[3 * H_ENC + k]);
            c[k] = fg * c[k] + ig * gg;
            h[k] = og * tanhf(c[k]);
        }
    }
    int seq = 0;
    for (int s = 0; s < NUM_SEQ; ++s)
        if (n >= seq_start_end[s * 2]) seq = s;
    #pragma unroll
    for (int k = 0; k < H_ENC; ++k) hbuf[n * H_DEC + k] = h[k];
    #pragma unroll
    for (int k = 0; k < NOISE_DIM; ++k) hbuf[n * H_DEC + H_ENC + k] = noise[seq * NOISE_DIM + k];
    #pragma unroll
    for (int k = 0; k < H_DEC; ++k) { cbuf[n * H_DEC + k] = 0.f; ctxbuf[n * H_DEC + k] = 0.f; }
    outbuf[n * 2 + 0] = 0.f;
    outbuf[n * 2 + 1] = 0.f;
    posbuf[n * 2 + 0] = obs_traj_pos[((size_t)(OBS_LEN - 1) * NAG + n) * 2 + 0];
    posbuf[n * 2 + 1] = obs_traj_pos[((size_t)(OBS_LEN - 1) * NAG + n) * 2 + 1];
}

// ---------------------------------------------------------------------------
// Decoder LSTM cell (H=16) per agent.
// ---------------------------------------------------------------------------
__global__ void dec_lstm_kernel(const float* __restrict__ W_dec, const float* __restrict__ b_dec,
                                const float* __restrict__ Wih_p, const float* __restrict__ Whh_p,
                                const float* __restrict__ bih_p, const float* __restrict__ bhh_p,
                                const float* __restrict__ ctxbuf, const float* __restrict__ outbuf,
                                float* __restrict__ hbuf, float* __restrict__ cbuf) {
    int n = blockIdx.x * blockDim.x + threadIdx.x;
    if (n >= NAG) return;
    float inp[H_DEC + 2];
    #pragma unroll
    for (int k = 0; k < H_DEC; ++k) inp[k] = ctxbuf[n * H_DEC + k];
    inp[H_DEC + 0] = outbuf[n * 2 + 0];
    inp[H_DEC + 1] = outbuf[n * 2 + 1];
    float e[EMB];
    #pragma unroll
    for (int m = 0; m < EMB; ++m) {
        float v = b_dec[m];
        #pragma unroll
        for (int j = 0; j < H_DEC + 2; ++j) v += W_dec[m * (H_DEC + 2) + j] * inp[j];
        e[m] = v > 0.f ? v : 0.f;
    }
    float h[H_DEC], c[H_DEC];
    #pragma unroll
    for (int k = 0; k < H_DEC; ++k) { h[k] = hbuf[n * H_DEC + k]; c[k] = cbuf[n * H_DEC + k]; }
    float g[4 * H_DEC];
    #pragma unroll
    for (int q = 0; q < 4 * H_DEC; ++q) {
        float v = bih_p[q] + bhh_p[q];
        #pragma unroll
        for (int m = 0; m < EMB; ++m) v += Wih_p[q * EMB + m] * e[m];
        #pragma unroll
        for (int k = 0; k < H_DEC; ++k) v += Whh_p[q * H_DEC + k] * h[k];
        g[q] = v;
    }
    #pragma unroll
    for (int k = 0; k < H_DEC; ++k) {
        float ig = sigm(g[k]), fg = sigm(g[H_DEC + k]);
        float gg = tanhf(g[2 * H_DEC + k]), og = sigm(g[3 * H_DEC + k]);
        float c2 = fg * c[k] + ig * gg;
        cbuf[n * H_DEC + k] = c2;
        hbuf[n * H_DEC + k] = og * tanhf(c2);
    }
}

// ---------------------------------------------------------------------------
// Social pool: one block per target agent i. 128 threads = 4 wave32.
// Transposed formulation keeps the whole MLP chain in registers:
//   h1^T  = Wm1 @ m_in^T   (A = Wm1, loop-invariant VGPR fragments)
//   pre^T = Wm2 @ relu(h1^T)  (A = Wm2; D'->B via one half-wave shfl_xor(16))
// m_in^T B-fragments are gathered from sibling lanes with __shfl (no LDS).
// Attention score comes straight out of the GEMM2 epilogue registers.
// LDS is used only for pre (cross-thread softmax context sum) + reductions.
// ---------------------------------------------------------------------------
__global__ __launch_bounds__(128) void pool_wmma_kernel(
    const float* __restrict__ pos, const float* __restrict__ hbuf,
    float* __restrict__ ctxbuf,
    const float* __restrict__ W_se, const float* __restrict__ b_se,
    const _Float16* __restrict__ Wm1h, const float* __restrict__ b_m1,
    const _Float16* __restrict__ Wm2h, const float* __restrict__ b_m2,
    const float* __restrict__ W_att, const float* __restrict__ b_att,
    const int* __restrict__ nei) {
    __shared__ float spre[128 * 16];       // pre, row-major f32 (softmax ctx sum)
    __shared__ float sE[128];              // exp(s - m) per row
    __shared__ float sWred[4], sEred[4];
    __shared__ float sP[8 * 16];           // partial ctx sums [group][col]
    __shared__ float s_runm, s_rund;
    __shared__ float s_ctx[16];
    __shared__ int   s_any;

    const int i    = blockIdx.x;
    const int t    = threadIdx.x;
    const int wave = t >> 5;
    const int lane = t & 31;
    const int hi   = lane >> 4;
    const int lx   = lane & 15;

    // ---- loop-invariant A operands in VGPRs (A layout: lane=M, vgpr=K)
    v16h aW1[4], aW2[2];
    #pragma unroll
    for (int nt = 0; nt < 4; ++nt) {
        #pragma unroll
        for (int p = 0; p < 16; ++p) {
            int k = p + hi * 16;
            aW1[nt][p] = Wm1h[(nt * 16 + lx) * 32 + k];
        }
    }
    #pragma unroll
    for (int kg = 0; kg < 2; ++kg) {
        #pragma unroll
        for (int p = 0; p < 16; ++p) {
            int k = p + hi * 16 + kg * 32;
            aW2[kg][p] = Wm2h[lx * 64 + k];
        }
    }
    float b1v[4][8], b2v[8], wattv[8];
    #pragma unroll
    for (int nt = 0; nt < 4; ++nt)
        #pragma unroll
        for (int r = 0; r < 8; ++r) b1v[nt][r] = b_m1[nt * 16 + r + 8 * hi];
    #pragma unroll
    for (int r = 0; r < 8; ++r) { b2v[r] = b_m2[r + 8 * hi]; wattv[r] = W_att[r + 8 * hi]; }
    const float batt = b_att[0];

    if (t == 0) { s_runm = -3.4e38f; s_rund = 0.0f; s_any = 0; }
    if (t < 16) s_ctx[t] = 0.0f;
    const float pix = pos[i * 2 + 0], piy = pos[i * 2 + 1];
    __syncthreads();

    for (int chunk = 0; chunk < NAG / 128; ++chunk) {
        const int j    = chunk * 128 + t;
        const int mval = nei[(size_t)i * NAG + j];   // early: overlap with GEMMs

        // ---- this thread's m_in row, packed f16 pairs (pk[q] = K 2q,2q+1)
        unsigned pk[16];
        {
            float dx = pix - pos[j * 2 + 0];
            float dy = piy - pos[j * 2 + 1];
            float rv[32];
            #pragma unroll
            for (int m = 0; m < EMB; ++m)
                rv[m] = b_se[m] + W_se[m * 2 + 0] * dx + W_se[m * 2 + 1] * dy;
            #pragma unroll
            for (int k = 0; k < H_DEC; ++k) rv[16 + k] = hbuf[j * H_DEC + k];
            #pragma unroll
            for (int q = 0; q < 16; ++q) pk[q] = pack_f16x2(rv[2 * q], rv[2 * q + 1]);
        }

        float sct[2];
        #pragma unroll
        for (int c = 0; c < 2; ++c) {
            const int ct = wave * 2 + c;           // this wave's column (row) tile
            // B-frag of m_in^T: lane = chunk row, K half by lane half
            VH bm;
            const int src = c * 16 + lx;           // source lane within wave
            #pragma unroll
            for (int q = 0; q < 8; ++q)
                bm.u[q] = (unsigned)__shfl((int)pk[q + 8 * hi], src, 32);

            // GEMM1': h1^T tiles (M = Wm1 rows), D' in registers
            float cn[4][8];
            #pragma unroll
            for (int nt = 0; nt < 4; ++nt) {
                v8f cc = {};
                cc = __builtin_amdgcn_wmma_f32_16x16x32_f16(false, aW1[nt], false, bm.v,
                                                            (short)0, cc, false, false);
                #pragma unroll
                for (int r = 0; r < 8; ++r) {
                    float v = cc[r] + b1v[nt][r];
                    cn[nt][r] = v > 0.f ? v : 0.f;
                }
            }

            // GEMM2': pre^T = Wm2 @ relu(h1^T); D'->B needs only shfl_xor(16)
            v8f c2 = {};
            #pragma unroll
            for (int kg = 0; kg < 2; ++kg) {
                unsigned p0[4], p1[4];
                #pragma unroll
                for (int q = 0; q < 4; ++q) {
                    p0[q] = pack_f16x2(cn[2 * kg][2 * q], cn[2 * kg][2 * q + 1]);
                    p1[q] = pack_f16x2(cn[2 * kg + 1][2 * q], cn[2 * kg + 1][2 * q + 1]);
                }
                VH bb;
                #pragma unroll
                for (int q = 0; q < 4; ++q) {
                    unsigned x0 = (unsigned)__shfl_xor((int)p0[q], 16, 32);
                    unsigned x1 = (unsigned)__shfl_xor((int)p1[q], 16, 32);
                    bb.u[q]     = hi ? x1 : p0[q];   // lower: K0..7 own, upper: K16..23 partner
                    bb.u[q + 4] = hi ? p1[q] : x0;   // lower: K8..15 partner, upper: K24..31 own
                }
                c2 = __builtin_amdgcn_wmma_f32_16x16x32_f16(false, aW2[kg], false, bb.v,
                                                            (short)0, c2, false, false);
            }

            // epilogue: bias+relu, store pre row-major, score partial from regs
            float sp = 0.f;
            #pragma unroll
            for (int r = 0; r < 8; ++r) {
                float v = c2[r] + b2v[r];
                v = v > 0.f ? v : 0.f;
                spre[(ct * 16 + lx) * 16 + 8 * hi + r] = v;   // contiguous 32B/lane
                sp += v * wattv[r];
            }
            sp += __shfl_xor(sp, 16, 32);          // combine the two n-halves
            sct[c] = sp;
        }

        // ---- this thread's own score (row = 32*wave + lane)
        float s = batt + (hi ? sct[1] : sct[0]);
        const bool masked = mval > 0;
        if (!masked) s = -1e9f;
        {
            float mw = wave_max_f32(s);
            if (lane == 0) sWred[wave] = mw;
            unsigned long long bm_ = __ballot(masked);
            if (lane == 0 && bm_ != 0ull) s_any = 1;
        }
        __syncthreads();                                               // 1

        const float old_m = s_runm;                      // uniform read
        float cmax = fmaxf(fmaxf(sWred[0], sWred[1]), fmaxf(sWred[2], sWred[3]));
        const float nm    = fmaxf(old_m, cmax);
        const float scale = __expf(old_m - nm);          // uniform
        const float e     = __expf(s - nm);
        sE[t] = e;
        {
            float ew = wave_sum_f32(e);
            if (lane == 0) sEred[wave] = ew;
        }
        __syncthreads();                                               // 2

        // partial ctx: thread (g,n) accumulates 16 rows of group g, column n
        {
            const int g = t >> 4, n = t & 15;
            float pacc = 0.f;
            #pragma unroll
            for (int r = 0; r < 16; ++r) {
                int q = g * 16 + r;
                pacc += sE[q] * spre[q * 16 + n];
            }
            sP[g * 16 + n] = pacc;
        }
        const float esum = sEred[0] + sEred[1] + sEred[2] + sEred[3];
        __syncthreads();                                               // 3

        if (t == 0) { s_rund = s_rund * scale + esum; s_runm = nm; }
        if (t < 16) {
            float acc = 0.f;
            #pragma unroll
            for (int g = 0; g < 8; ++g) acc += sP[g * 16 + t];
            s_ctx[t] = s_ctx[t] * scale + acc;
        }
        __syncthreads();                                               // 4
    }

    if (t < 16) {
        float v = s_any ? (s_ctx[t] / s_rund) : 0.f;
        ctxbuf[i * H_DEC + t] = v;
    }
}

// ---------------------------------------------------------------------------
// Output head: out = tanh((ctx+h) @ W_out^T + b_out) * 4.4 ; pos += out
// ---------------------------------------------------------------------------
__global__ void dec_out_kernel(const float* __restrict__ W_out, const float* __restrict__ b_out,
                               const float* __restrict__ ctxbuf, const float* __restrict__ hbuf,
                               float* __restrict__ outbuf, float* __restrict__ posbuf,
                               float* __restrict__ pred_out) {
    int n = blockIdx.x * blockDim.x + threadIdx.x;
    if (n >= NAG) return;
    float o[2];
    #pragma unroll
    for (int d = 0; d < 2; ++d) {
        float v = b_out[d];
        #pragma unroll
        for (int k = 0; k < H_DEC; ++k)
            v += (ctxbuf[n * H_DEC + k] + hbuf[n * H_DEC + k]) * W_out[d * H_DEC + k];
        o[d] = tanhf(v) * 4.4f;
        pred_out[n * 2 + d] = o[d];
        outbuf[n * 2 + d]   = o[d];
    }
    posbuf[n * 2 + 0] += o[0];
    posbuf[n * 2 + 1] += o[1];
}

extern "C" void kernel_launch(void* const* d_in, const int* in_sizes, int n_in,
                              void* d_out, int out_size, void* d_ws, size_t ws_size,
                              hipStream_t stream) {
    (void)in_sizes; (void)n_in; (void)out_size; (void)ws_size;
    const float* traj_rel     = (const float*)d_in[0];
    const float* obs_traj_pos = (const float*)d_in[1];
    /* d_in[2] pred_traj_gt_pos unused by reference */
    const float* h0    = (const float*)d_in[3];
    const float* c0    = (const float*)d_in[4];
    const float* noise = (const float*)d_in[5];
    const float* W_enc = (const float*)d_in[6];  const float* b_enc = (const float*)d_in[7];
    const float* W_dec = (const float*)d_in[8];  const float* b_dec = (const float*)d_in[9];
    const float* Wih_t = (const float*)d_in[10]; const float* Whh_t = (const float*)d_in[11];
    const float* bih_t = (const float*)d_in[12]; const float* bhh_t = (const float*)d_in[13];
    const float* Wih_p = (const float*)d_in[14]; const float* Whh_p = (const float*)d_in[15];
    const float* bih_p = (const float*)d_in[16]; const float* bhh_p = (const float*)d_in[17];
    const float* W_out = (const float*)d_in[18]; const float* b_out = (const float*)d_in[19];
    const float* W_se  = (const float*)d_in[20]; const float* b_se  = (const float*)d_in[21];
    const float* W_m1  = (const float*)d_in[22]; const float* b_m1  = (const float*)d_in[23];
    const float* W_m2  = (const float*)d_in[24]; const float* b_m2  = (const float*)d_in[25];
    const float* W_att = (const float*)d_in[26]; const float* b_att = (const float*)d_in[27];
    const int* seq_start_end = (const int*)d_in[28];
    const int* nei_index     = (const int*)d_in[29];
    /* d_in[30] nei_num_index unused by reference */

    char* ws = (char*)d_ws;
    float* hbuf   = (float*)ws; ws += NAG * H_DEC * sizeof(float);
    float* cbuf   = (float*)ws; ws += NAG * H_DEC * sizeof(float);
    float* ctxbuf = (float*)ws; ws += NAG * H_DEC * sizeof(float);
    float* outbuf = (float*)ws; ws += NAG * 2 * sizeof(float);
    float* posbuf = (float*)ws; ws += NAG * 2 * sizeof(float);
    _Float16* Wm1h = (_Float16*)ws; ws += 64 * 32 * sizeof(_Float16);
    _Float16* Wm2h = (_Float16*)ws; ws += 16 * 64 * sizeof(_Float16);
    float* outp = (float*)d_out;

    conv_weights_kernel<<<16, 128, 0, stream>>>(W_m1, W_m2, Wm1h, Wm2h);
    encoder_kernel<<<4, 128, 0, stream>>>(traj_rel, obs_traj_pos, h0, c0, noise,
                                          W_enc, b_enc, Wih_t, Whh_t, bih_t, bhh_t,
                                          seq_start_end,
                                          hbuf, cbuf, ctxbuf, outbuf, posbuf);
    for (int t = 0; t < PRED_LEN; ++t) {
        dec_lstm_kernel<<<4, 128, 0, stream>>>(W_dec, b_dec, Wih_p, Whh_p, bih_p, bhh_p,
                                               ctxbuf, outbuf, hbuf, cbuf);
        pool_wmma_kernel<<<NAG, 128, 0, stream>>>(posbuf, hbuf, ctxbuf,
                                                  W_se, b_se, Wm1h, b_m1, Wm2h, b_m2,
                                                  W_att, b_att,
                                                  nei_index + (size_t)t * NAG * NAG);
        dec_out_kernel<<<4, 128, 0, stream>>>(W_out, b_out, ctxbuf, hbuf, outbuf, posbuf,
                                              outp + (size_t)t * NAG * 2);
    }
}
